// ACTClassifier_89189290869298
// MI455X (gfx1250) — matrix operand order repaired and checked by reading, not compiled, not run on hardware
//
#include <hip/hip_runtime.h>
#include <hip/hip_bf16.h>

// ---------------------------------------------------------------------------
// ACT classifier (GRU ponder loop) for MI455X / gfx1250, wave32 + WMMA bf16.
// B=65536, D=512, H=256, C=100, T=10, eps=0.01
//
// Persistent batch-tile design: one 256-thread workgroup owns 64 batch rows
// for all T ponder steps. x_proj (bf16), double-buffered h (bf16) and the
// fp32 accumulator live in LDS for the whole kernel; weights are bf16 in the
// workspace and stay L2-resident (~1.2 MB vs 192 MB L2). All large outputs
// are streamed with non-temporal stores so the ~1 GB output stream does not
// evict the hot weights from L2.
// ---------------------------------------------------------------------------

typedef __bf16 bf16_t;
typedef __attribute__((ext_vector_type(16))) __bf16 v16bf;
typedef __attribute__((ext_vector_type(8)))  float  v8f;

#define ACT_B 65536
#define ACT_D 512
#define ACT_H 256
#define ACT_C 100
#define ACT_T 10
#define ACT_TB 64          // batch rows per workgroup
#define ACT_H3 768         // 3*H
#define ACT_THRESH 0.99f   // 1 - eps

__device__ __forceinline__ float act_sigmoid(float x) {
    return 1.0f / (1.0f + __expf(-x));
}

__device__ __forceinline__ v8f wmma_bf16(v16bf a, v16bf b, v8f c) {
    // v_wmma_f32_16x16x32_bf16 : D = A(16x32) * B(32x16) + C(16x16 f32)
    return __builtin_amdgcn_wmma_f32_16x16x32_bf16(
        /*neg_a=*/false, a, /*neg_b=*/false, b,
        /*c_mod=*/(short)0, c, /*reuse_a=*/false, /*reuse_b=*/false);
}

// A-matrix loader (16x32 bf16 tile, row-major source with 'stride' elements).
// Lane L holds row M=L%16; lo-lanes K 0..15, hi-lanes K 16..31 (matches load_B);
// packed 2/VGPR -> one contiguous 32B chunk per lane.
__device__ __forceinline__ v16bf load_A(const bf16_t* base, int stride,
                                        int lane, int k0) {
    const int m  = lane & 15;
    const int hi = lane >> 4;
    const bf16_t* p = base + (size_t)m * stride + k0 + hi * 16;
    v16bf a;
#pragma unroll
    for (int i = 0; i < 16; ++i) a[i] = p[i];         // 32B contiguous
    return a;
}

// B-matrix loader for B = W^T: column N of B == row N of row-major W[N][K].
// Lane L holds column N = n0 + L%16; lo-lanes K k0..k0+15, hi-lanes +16.
// Rows >= nLimit are zero-padded (used for C=100 -> 112 padding).
__device__ __forceinline__ v16bf load_B(const bf16_t* W, int K, int n0,
                                        int lane, int k0, int nLimit) {
    const int n  = n0 + (lane & 15);
    const int kk = k0 + ((lane >> 4) << 4);
    v16bf b;
    if (n < nLimit) {
        const bf16_t* p = W + (size_t)n * K + kk;
#pragma unroll
        for (int i = 0; i < 16; ++i) b[i] = p[i];     // 32B contiguous
    } else {
#pragma unroll
        for (int i = 0; i < 16; ++i) b[i] = (bf16_t)0.0f;
    }
    return b;
}

// ---------------------------------------------------------------------------
// small helper kernels
// ---------------------------------------------------------------------------
__global__ void act_cvt_f32_bf16(const float* __restrict__ s,
                                 bf16_t* __restrict__ d, int n) {
    int i = blockIdx.x * 256 + threadIdx.x;
    if (i < n) d[i] = (bf16_t)s[i];
}

__global__ void act_zero_one(float* p) { *p = 0.0f; }

// ---------------------------------------------------------------------------
// LDS layout (bytes):
//   [0)        x_proj bf16  64x768                       =  98304
//   [98304)    h buf 0 bf16 64x256                       =  32768
//   [131072)   h buf 1 bf16 64x256                       =  32768
//   [163840)   acc f32 64x256 (alias: x-stage bf16)      =  65536
//   [229376)   halt reduce f32 4x64                      =   1024
//   [230400)   cum/rem/nup/stepw f32 64 each             =   1024
//   [231424)   b_hh f32 768                              =   3072
//   [234496)   b_ro f32 112 (zero padded)                =    448
//   [234944)   W_halt f32 256                            =   1024
// total 235968 <= 320KB  -> 1 workgroup (8 waves, 2/SIMD32) per WGP
// ---------------------------------------------------------------------------
#define LDS_BYTES 235968

__global__ void __launch_bounds__(256, 1)
act_main_kernel(const float* __restrict__ x,
                const float* __restrict__ b_ih,
                const float* __restrict__ b_hh,
                const float* __restrict__ W_halt,
                const float* __restrict__ b_halt,
                const float* __restrict__ b_ro,
                const bf16_t* __restrict__ Wih_bf,
                const bf16_t* __restrict__ Whh_bf,
                const bf16_t* __restrict__ Wro_bf,
                float* __restrict__ out_logits,
                float* __restrict__ out_ponder,
                float* __restrict__ out_nup,
                float* __restrict__ out_hseq,
                float* __restrict__ out_haltseq,
                float* __restrict__ out_logseq) {
    extern __shared__ char smem[];
    bf16_t* sXP    = (bf16_t*)(smem);                   // 64x768 bf16
    bf16_t* sH0    = (bf16_t*)(smem + 98304);
    bf16_t* sH1    = (bf16_t*)(smem + 131072);
    float*  sAcc   = (float*) (smem + 163840);          // 64x256 f32
    bf16_t* sXst   = (bf16_t*)(smem + 163840);          // alias: 64x512 bf16
    float*  sRed   = (float*) (smem + 229376);          // 4x64
    float*  sCum   = (float*) (smem + 230400);
    float*  sRem   = sCum + 64;
    float*  sNup   = sRem + 64;
    float*  sStepw = sNup + 64;
    float*  sBhh   = (float*) (smem + 231424);          // 768
    float*  sBro   = (float*) (smem + 234496);          // 112
    float*  sWh    = (float*) (smem + 234944);          // 256

    const int tid  = threadIdx.x;
    const int lane = tid & 31;
    const int wave = tid >> 5;          // 0..7
    const int hi   = lane >> 4;
    const int b0   = blockIdx.x * ACT_TB;

    // ---- Phase 0: stage x tile (NT fp32 load -> bf16 LDS) + small vectors -
    {
        const float* xt = x + (size_t)b0 * ACT_D;
        for (int idx = tid; idx < ACT_TB * ACT_D; idx += 256)
            sXst[idx] = (bf16_t)__builtin_nontemporal_load(xt + idx);
        for (int i = tid; i < ACT_H3; i += 256) sBhh[i] = b_hh[i];
        if (tid < 112) sBro[tid] = (tid < ACT_C) ? b_ro[tid] : 0.0f;
        if (tid < 256) sWh[tid] = W_halt[tid];
    }
    __syncthreads();

    // ---- Phase 1: x_proj = x @ W_ih^T + b_ih  (WMMA, K=512) --------------
    {
        const int msub = wave & 3;                       // constant per wave
        const bf16_t* Abase = sXst + (size_t)msub * 16 * ACT_D;
        for (int tile = wave; tile < 4 * 48; tile += 8) {
            const int nt = tile >> 2;                    // 0..47
            v8f c = {};
#pragma unroll
            for (int kc = 0; kc < ACT_D / 32; ++kc) {
                v16bf a = load_A(Abase, ACT_D, lane, kc * 32);
                v16bf b = load_B(Wih_bf, ACT_D, nt * 16, lane, kc * 32, ACT_H3);
                c = wmma_bf16(a, b, c);
            }
            const int n = nt * 16 + (lane & 15);
            const float bi = b_ih[n];
#pragma unroll
            for (int v = 0; v < 8; ++v) {
                const int row = msub * 16 + v + 8 * hi;
                sXP[(size_t)row * ACT_H3 + n] = (bf16_t)(c[v] + bi);
            }
        }
    }
    __syncthreads();

    // ---- Phase 2: init h=0, acc=0, ACT state ------------------------------
    for (int idx = tid; idx < ACT_TB * ACT_H; idx += 256) {
        sAcc[idx] = 0.0f;
        sH0[idx]  = (bf16_t)0.0f;
    }
    if (tid < 64) { sCum[tid] = 0.0f; sRem[tid] = 0.0f; sNup[tid] = 0.0f; }
    __syncthreads();

    bf16_t* hc  = sH0;   // current h
    bf16_t* hnx = sH1;   // next h

    for (int t = 0; t < ACT_T; ++t) {
        // ---- Phase A: h_proj (r/z/n) + fused GRU gating -------------------
        {
            const int msub = wave >> 1;                 // 0..3
            const int jset = wave & 1;                  // half of H cols
            const bf16_t* Abase = hc + (size_t)msub * 16 * ACT_H;
            v16bf Ah[8];
#pragma unroll
            for (int kc = 0; kc < 8; ++kc)
                Ah[kc] = load_A(Abase, ACT_H, lane, kc * 32);

            for (int jb = jset * 8; jb < jset * 8 + 8; ++jb) {
                v8f cr = {}, cz = {}, cn = {};
#pragma unroll
                for (int kc = 0; kc < 8; ++kc)
                    cr = wmma_bf16(Ah[kc],
                        load_B(Whh_bf, ACT_H, jb * 16, lane, kc * 32, ACT_H3), cr);
#pragma unroll
                for (int kc = 0; kc < 8; ++kc)
                    cz = wmma_bf16(Ah[kc],
                        load_B(Whh_bf, ACT_H, 256 + jb * 16, lane, kc * 32, ACT_H3), cz);
#pragma unroll
                for (int kc = 0; kc < 8; ++kc)
                    cn = wmma_bf16(Ah[kc],
                        load_B(Whh_bf, ACT_H, 512 + jb * 16, lane, kc * 32, ACT_H3), cn);

                const int nj  = jb * 16 + (lane & 15);   // col in H
                const float bhr = sBhh[nj];
                const float bhz = sBhh[256 + nj];
                const float bhn = sBhh[512 + nj];
#pragma unroll
                for (int v = 0; v < 8; ++v) {
                    const int row = msub * 16 + v + 8 * hi;
                    const size_t xo = (size_t)row * ACT_H3 + nj;
                    const float xr = (float)sXP[xo];
                    const float xz = (float)sXP[xo + 256];
                    const float xn = (float)sXP[xo + 512];
                    const float hold = (float)hc[(size_t)row * ACT_H + nj];
                    const float r = act_sigmoid(xr + cr[v] + bhr);
                    const float z = act_sigmoid(xz + cz[v] + bhz);
                    const float nn = tanhf(xn + r * (cn[v] + bhn));
                    const float hv = (1.0f - z) * nn + z * hold;
                    hnx[(size_t)row * ACT_H + nj] = (bf16_t)hv;
                    __builtin_nontemporal_store(hv,
                        out_hseq + ((size_t)(b0 + row) * ACT_T + t) * ACT_H + nj);
                }
            }
        }
        __syncthreads();

        // ---- Phase B: halting GEMV + ACT state ----------------------------
        {
            const int r = tid & 63, q = tid >> 6;
            float s = 0.0f;
#pragma unroll 4
            for (int k = q * 64; k < q * 64 + 64; ++k)
                s += (float)hnx[(size_t)r * ACT_H + k] * sWh[k];
            sRed[q * 64 + r] = s;
        }
        __syncthreads();
        if (tid < 64) {
            const float s = sRed[tid] + sRed[64 + tid] + sRed[128 + tid] +
                            sRed[192 + tid] + b_halt[0];
            const float halt = act_sigmoid(s);
            const float cum  = sCum[tid];
            const float still = (cum < ACT_THRESH) ? 1.0f : 0.0f;
            const float nh = halt * still;
            const float would = (cum + nh > ACT_THRESH) ? 1.0f : 0.0f;
            const float remainder = (1.0f - cum) * would * still;
            const float sw = nh * (1.0f - would) + remainder;
            sCum[tid]  = cum + sw;
            sRem[tid] += remainder;
            sNup[tid] += still;
            sStepw[tid] = sw;
            __builtin_nontemporal_store(halt,
                out_haltseq + (size_t)(b0 + tid) * ACT_T + t);
        }
        __syncthreads();

        // ---- Phase C: acc += step_w * h_new -------------------------------
        for (int idx = tid; idx < ACT_TB * ACT_H; idx += 256) {
            const int r = idx >> 8;
            sAcc[idx] += sStepw[r] * (float)hnx[idx];
        }

        // ---- Phase D: logits_t = h_new @ W_ro^T + b_ro (WMMA) -------------
        for (int tile = wave; tile < 4 * 7; tile += 8) {
            const int ms  = tile / 7;
            const int ntc = tile % 7;
            const bf16_t* Abase = hnx + (size_t)ms * 16 * ACT_H;
            v8f c = {};
#pragma unroll
            for (int kc = 0; kc < 8; ++kc)
                c = wmma_bf16(load_A(Abase, ACT_H, lane, kc * 32),
                              load_B(Wro_bf, ACT_H, ntc * 16, lane, kc * 32, ACT_C),
                              c);
            const int n = ntc * 16 + (lane & 15);
            if (n < ACT_C) {
                const float br = sBro[n];
#pragma unroll
                for (int v = 0; v < 8; ++v) {
                    const int row = ms * 16 + v + 8 * hi;
                    __builtin_nontemporal_store(c[v] + br,
                        out_logseq + ((size_t)(b0 + row) * ACT_T + t) * ACT_C + n);
                }
            }
        }
        __syncthreads();

        bf16_t* tmp = hc; hc = hnx; hnx = tmp;   // swap h buffers
    }

    // ---- Final: logits = acc @ W_ro^T + b_ro ------------------------------
    for (int idx = tid; idx < ACT_TB * ACT_H; idx += 256)
        sH0[idx] = (bf16_t)sAcc[idx];
    __syncthreads();

    for (int tile = wave; tile < 4 * 7; tile += 8) {
        const int ms  = tile / 7;
        const int ntc = tile % 7;
        const bf16_t* Abase = sH0 + (size_t)ms * 16 * ACT_H;
        v8f c = {};
#pragma unroll
        for (int kc = 0; kc < 8; ++kc)
            c = wmma_bf16(load_A(Abase, ACT_H, lane, kc * 32),
                          load_B(Wro_bf, ACT_H, ntc * 16, lane, kc * 32, ACT_C),
                          c);
        const int n = ntc * 16 + (lane & 15);
        if (n < ACT_C) {
            const float br = sBro[n];
#pragma unroll
            for (int v = 0; v < 8; ++v) {
                const int row = ms * 16 + v + 8 * hi;
                __builtin_nontemporal_store(c[v] + br,
                    out_logits + (size_t)(b0 + row) * ACT_C + n);
            }
        }
    }

    if (tid < 64)
        __builtin_nontemporal_store(sNup[tid], out_nup + b0 + tid);
    if (tid == 0) {
        float s = 0.0f;
        for (int i = 0; i < 64; ++i) s += sNup[i] + sRem[i];
        atomicAdd(out_ponder, s / (float)ACT_B);
    }
}

// ---------------------------------------------------------------------------
extern "C" void kernel_launch(void* const* d_in, const int* in_sizes, int n_in,
                              void* d_out, int out_size, void* d_ws, size_t ws_size,
                              hipStream_t stream) {
    const float* x      = (const float*)d_in[0];
    const float* W_ih   = (const float*)d_in[1];
    const float* W_hh   = (const float*)d_in[2];
    const float* b_ih   = (const float*)d_in[3];
    const float* b_hh   = (const float*)d_in[4];
    const float* W_halt = (const float*)d_in[5];
    const float* b_halt = (const float*)d_in[6];
    const float* W_ro   = (const float*)d_in[7];
    const float* b_ro   = (const float*)d_in[8];

    // bf16 weight copies in workspace (L2-resident, ~1.2 MB total)
    bf16_t* ws     = (bf16_t*)d_ws;
    bf16_t* Wih_bf = ws;                    // 768*512 = 393216
    bf16_t* Whh_bf = ws + 393216;           // 768*256 = 196608
    bf16_t* Wro_bf = ws + 589824;           // 100*256 = 25600

    // output layout: logits | ponder | n_updates | h_seq | halt_seq | logits_seq
    float* out         = (float*)d_out;
    float* out_logits  = out;                             // B*C
    float* out_ponder  = out + (size_t)ACT_B * ACT_C;     // 1
    float* out_nup     = out_ponder + 1;                  // B
    float* out_hseq    = out_nup + ACT_B;                 // B*T*H
    float* out_haltseq = out_hseq + (size_t)ACT_B * ACT_T * ACT_H;  // B*T
    float* out_logseq  = out_haltseq + (size_t)ACT_B * ACT_T;       // B*T*C

    act_cvt_f32_bf16<<<(393216 + 255) / 256, 256, 0, stream>>>(W_ih, Wih_bf, 393216);
    act_cvt_f32_bf16<<<(196608 + 255) / 256, 256, 0, stream>>>(W_hh, Whh_bf, 196608);
    act_cvt_f32_bf16<<<(25600  + 255) / 256, 256, 0, stream>>>(W_ro, Wro_bf, 25600);
    act_zero_one<<<1, 1, 0, stream>>>(out_ponder);

    (void)hipFuncSetAttribute((const void*)act_main_kernel,
                              hipFuncAttributeMaxDynamicSharedMemorySize,
                              LDS_BYTES);

    act_main_kernel<<<ACT_B / ACT_TB, 256, LDS_BYTES, stream>>>(
        x, b_ih, b_hh, W_halt, b_halt, b_ro,
        Wih_bf, Whh_bf, Wro_bf,
        out_logits, out_ponder, out_nup, out_hseq, out_haltseq, out_logseq);
}
